// Linear_23201413333001
// MI455X (gfx1250) — matrix operand order repaired and checked by reading, not compiled, not run on hardware
//
#include <hip/hip_runtime.h>

// ---------------------------------------------------------------------------
// Problem dims
// ---------------------------------------------------------------------------
#define M_TOK   8192
#define IN_DIM  4096
#define OUT_DIM 4096
#define RANK    16

// GEMM tiling
#define BM 128
#define BN 128
#define BK 32
#define LDS_STRIDE 40                    // halves per row (BK + 8 pad, 16B aligned rows)
#define TILE_BYTES (BM * LDS_STRIDE * 2) // 10240 B per tile array
#define AHI_OFF 0
#define ALO_OFF (TILE_BYTES)
#define BHI_OFF (2 * TILE_BYTES)
#define BLO_OFF (3 * TILE_BYTES)
#define BUF_BYTES (4 * TILE_BYTES)       // 40960 B per stage
#define SMEM_BYTES (2 * BUF_BYTES)       // 81920 B, ping-pong

typedef __attribute__((ext_vector_type(16))) __bf16 bf16x16;
typedef __attribute__((ext_vector_type(8)))  __bf16 bf16x8;
typedef __attribute__((ext_vector_type(4)))  __bf16 bf16x4;
typedef __attribute__((ext_vector_type(8)))  float  floatx8;
typedef __attribute__((ext_vector_type(4)))  float  floatx4;

static __device__ __forceinline__ __bf16 u16_as_bf16(unsigned short u) {
    union { unsigned short u; __bf16 b; } v; v.u = u; return v.b;
}

// Split fp32 into hi/lo bf16 via bit manipulation (no bf16 arithmetic).
static __device__ __forceinline__ void split_f32(float x, __bf16& hi, __bf16& lo) {
    unsigned xb = __float_as_uint(x);
    hi = u16_as_bf16((unsigned short)(xb >> 16));          // truncated high part
    float hf = __uint_as_float(xb & 0xffff0000u);
    unsigned rb = __float_as_uint(x - hf);                 // exact residual
    lo = u16_as_bf16((unsigned short)(rb >> 16));
}

// Build a 16x32 bf16 fragment from two contiguous 16B LDS chunks.
static __device__ __forceinline__ bf16x16 ld_frag(const __bf16* p0, const __bf16* p1) {
    bf16x8 a = *(const bf16x8*)p0;
    bf16x8 b = *(const bf16x8*)p1;
    bf16x16 r;
#pragma unroll
    for (int i = 0; i < 8; ++i) { r[i] = a[i]; r[i + 8] = b[i]; }
    return r;
}

// ---------------------------------------------------------------------------
// Kernel 0: one-time split of W into bf16 hi/lo planes (W reused 64x by GEMM).
// ---------------------------------------------------------------------------
__global__ __launch_bounds__(256)
void split_w(const float* __restrict__ W,
             __bf16* __restrict__ Whi, __bf16* __restrict__ Wlo)
{
    size_t i = ((size_t)blockIdx.x * 256 + threadIdx.x) * 4;
    floatx4 v = *(const floatx4*)(W + i);
    bf16x4 h4, l4;
#pragma unroll
    for (int e = 0; e < 4; ++e) {
        __bf16 h, l;
        split_f32(v[e], h, l);
        h4[e] = h; l4[e] = l;
    }
    *(bf16x4*)(Whi + i) = h4;
    *(bf16x4*)(Wlo + i) = l4;
}

// ---------------------------------------------------------------------------
// Kernel 1: out = data @ W^T via split-bf16 (hi*hi + hi*lo + lo*hi) WMMA.
// Ping-pong LDS; A staged via load+split+ds_store, B staged via
// global_load_async_to_lds_b128 (ASYNCcnt) from the pre-split W planes.
// Grid: (OUT_DIM/BN, M_TOK/BM), Block: 256 threads (8 wave32).
// ---------------------------------------------------------------------------
__global__ __launch_bounds__(256)
void gemm_bf16x3(const float* __restrict__ A,        // [M_TOK, IN_DIM] fp32
                 const __bf16* __restrict__ Whi,     // [OUT_DIM, IN_DIM]
                 const __bf16* __restrict__ Wlo,     // [OUT_DIM, IN_DIM]
                 float* __restrict__ C)              // [M_TOK, OUT_DIM]
{
    extern __shared__ __align__(16) char smem[];

    const int tid   = threadIdx.x;
    const int lane  = tid & 31;
    const int wave  = tid >> 5;
    const int waveM = wave & 1;      // 0..1 -> 64-row slab
    const int waveN = wave >> 1;     // 0..3 -> 32-col slab
    const int blockM = blockIdx.y * BM;
    const int blockN = blockIdx.x * BN;
    const int lan15 = lane & 15;
    const int kgrp  = lane >> 4;     // 0/1: which K-half this lane owns

    floatx8 acc[4][2];
#pragma unroll
    for (int i = 0; i < 4; ++i)
#pragma unroll
        for (int j = 0; j < 2; ++j)
#pragma unroll
            for (int e = 0; e < 8; ++e) acc[i][j][e] = 0.0f;

    auto stage = [&](int p, int k0) {
        char* base = smem + p * BUF_BYTES;
        __bf16* sAhi = (__bf16*)(base + AHI_OFF);
        __bf16* sAlo = (__bf16*)(base + ALO_OFF);
        // ---- A: global fp32 -> split bf16 -> ds_store ----
#pragma unroll
        for (int i = 0; i < 4; ++i) {
            int c    = tid + 256 * i;      // 1024 float4 chunks
            int row  = c >> 3;             // 0..127
            int col4 = c & 7;              // 4-float chunk within BK
            floatx4 va = *(const floatx4*)(A + (size_t)(blockM + row) * IN_DIM + k0 + col4 * 4);
            bf16x4 ah, al;
#pragma unroll
            for (int e = 0; e < 4; ++e) {
                __bf16 h, l;
                split_f32(va[e], h, l);
                ah[e] = h; al[e] = l;
            }
            *(bf16x4*)(&sAhi[row * LDS_STRIDE + col4 * 4]) = ah;
            *(bf16x4*)(&sAlo[row * LDS_STRIDE + col4 * 4]) = al;
        }
        // ---- B: pre-split bf16 planes -> async copy straight into LDS ----
#pragma unroll
        for (int i = 0; i < 2; ++i) {
            int c    = tid + 256 * i;      // 512 b128 chunks per plane
            int row  = c >> 2;             // 0..127
            int col8 = c & 3;              // 8-half chunk within BK
            unsigned rel  = (unsigned)(row * LDS_STRIDE + col8 * 8) * 2u;
            unsigned ldsH = (unsigned)(uintptr_t)(base + BHI_OFF + rel);
            unsigned ldsL = (unsigned)(uintptr_t)(base + BLO_OFF + rel);
            const __bf16* gH = Whi + (size_t)(blockN + row) * IN_DIM + k0 + col8 * 8;
            const __bf16* gL = Wlo + (size_t)(blockN + row) * IN_DIM + k0 + col8 * 8;
            asm volatile("global_load_async_to_lds_b128 %0, %1, off"
                         :: "v"(ldsH), "v"(gH) : "memory");
            asm volatile("global_load_async_to_lds_b128 %0, %1, off"
                         :: "v"(ldsL), "v"(gL) : "memory");
        }
    };

    auto compute = [&](int p) {
        char* base = smem + p * BUF_BYTES;
        const __bf16* sAhi = (const __bf16*)(base + AHI_OFF);
        const __bf16* sAlo = (const __bf16*)(base + ALO_OFF);
        const __bf16* sBhi = (const __bf16*)(base + BHI_OFF);
        const __bf16* sBlo = (const __bf16*)(base + BLO_OFF);

        bf16x16 afh[4], afl[4], bfh[2], bfl[2];
#pragma unroll
        for (int i = 0; i < 4; ++i) {
            int mrow = waveM * 64 + i * 16 + lan15;
            afh[i] = ld_frag(&sAhi[mrow * LDS_STRIDE + kgrp * 8],
                             &sAhi[mrow * LDS_STRIDE + 16 + kgrp * 8]);
            afl[i] = ld_frag(&sAlo[mrow * LDS_STRIDE + kgrp * 8],
                             &sAlo[mrow * LDS_STRIDE + 16 + kgrp * 8]);
        }
#pragma unroll
        for (int j = 0; j < 2; ++j) {
            int ncol = waveN * 32 + j * 16 + lan15;
            const __bf16* ph = &sBhi[ncol * LDS_STRIDE + kgrp * 8];
            const __bf16* pl = &sBlo[ncol * LDS_STRIDE + kgrp * 8];
            bfh[j] = ld_frag(ph, ph + 16);
            bfl[j] = ld_frag(pl, pl + 16);
        }
#pragma unroll
        for (int i = 0; i < 4; ++i) {
#pragma unroll
            for (int j = 0; j < 2; ++j) {
                acc[i][j] = __builtin_amdgcn_wmma_f32_16x16x32_bf16(
                    false, afh[i], false, bfh[j], (short)0, acc[i][j], false, false);
                acc[i][j] = __builtin_amdgcn_wmma_f32_16x16x32_bf16(
                    false, afh[i], false, bfl[j], (short)0, acc[i][j], false, false);
                acc[i][j] = __builtin_amdgcn_wmma_f32_16x16x32_bf16(
                    false, afl[i], false, bfh[j], (short)0, acc[i][j], false, false);
            }
        }
    };

    // prologue: fill buffer 0
    stage(0, 0);
    asm volatile("s_wait_asynccnt 0" ::: "memory");
    __syncthreads();

    for (int it = 0; it < IN_DIM / BK; ++it) {
        int p = it & 1;
        int knext = (it + 1) * BK;
        if (knext < IN_DIM) {
            __builtin_prefetch(A + (size_t)(blockM + (tid >> 1)) * IN_DIM + knext + BK, 0, 1);
            stage(1 - p, knext);          // overlap with WMMAs below
        }
        compute(p);
        asm volatile("s_wait_asynccnt 0" ::: "memory");
        __syncthreads();
    }

    // ---- epilogue: store C per documented 16x16 f32 D layout ----
#pragma unroll
    for (int i = 0; i < 4; ++i) {
#pragma unroll
        for (int j = 0; j < 2; ++j) {
            int mBase = blockM + waveM * 64 + i * 16 + kgrp * 8;
            int nCol  = blockN + waveN * 32 + j * 16 + lan15;
#pragma unroll
            for (int e = 0; e < 8; ++e) {
                C[(size_t)(mBase + e) * OUT_DIM + nCol] = acc[i][j][e];
            }
        }
    }
}

// ---------------------------------------------------------------------------
// Kernel 2: rank-16 projections T[m, r] = data[m,:] @ Aseg^T  (d_vec folded
// for VeRA segment). One block per token row.
// ---------------------------------------------------------------------------
__global__ __launch_bounds__(256)
void adapter_proj(const float* __restrict__ data,
                  const float* __restrict__ l1a, const float* __restrict__ l2a,
                  const float* __restrict__ va,  const float* __restrict__ da,
                  const float* __restrict__ d_vec,
                  float* __restrict__ T)         // [M_TOK, RANK]
{
    const int row = blockIdx.x;
    const int seg = row >> 11;                   // 2048-token segments
    const float* Amat = (seg == 0) ? l1a : (seg == 1) ? l2a : (seg == 2) ? va : da;
    const int t = threadIdx.x;

    float acc[RANK];
#pragma unroll
    for (int r = 0; r < RANK; ++r) acc[r] = 0.0f;

#pragma unroll 4
    for (int i = 0; i < IN_DIM / 256; ++i) {
        int k = t + 256 * i;
        float x = data[(size_t)row * IN_DIM + k];
#pragma unroll
        for (int r = 0; r < RANK; ++r)
            acc[r] = fmaf(x, Amat[r * IN_DIM + k], acc[r]);
    }

    __shared__ float red[RANK][256];
#pragma unroll
    for (int r = 0; r < RANK; ++r) red[r][t] = acc[r];
    __syncthreads();
    for (int s = 128; s > 0; s >>= 1) {
        if (t < s) {
#pragma unroll
            for (int r = 0; r < RANK; ++r) red[r][t] += red[r][t + s];
        }
        __syncthreads();
    }
    if (t < RANK) {
        float v = red[t][0];
        if (seg == 2) v *= d_vec[t];             // VeRA diagonal (scaling = 1.0)
        T[(size_t)row * RANK + t] = v;
    }
}

// ---------------------------------------------------------------------------
// Kernel 3: DoRA mns[o] = magnitude[o] / || W[o,:] + 2*(dora_b @ dora_a)[o,:] ||
// One block per output row.
// ---------------------------------------------------------------------------
__global__ __launch_bounds__(256)
void dora_norm(const float* __restrict__ W,
               const float* __restrict__ da,   // [16, IN_DIM]
               const float* __restrict__ db,   // [OUT_DIM, 16]
               const float* __restrict__ mag,
               float* __restrict__ mns)
{
    const int o = blockIdx.x;
    const int t = threadIdx.x;

    __shared__ float sdb[RANK];
    if (t < RANK) sdb[t] = 2.0f * db[(size_t)o * RANK + t];   // DORA_SCALING folded
    __syncthreads();

    float b[RANK];
#pragma unroll
    for (int r = 0; r < RANK; ++r) b[r] = sdb[r];

    float acc = 0.0f;
#pragma unroll 4
    for (int i = 0; i < IN_DIM / 256; ++i) {
        int k = t + 256 * i;
        float w = W[(size_t)o * IN_DIM + k];
#pragma unroll
        for (int r = 0; r < RANK; ++r)
            w = fmaf(b[r], da[r * IN_DIM + k], w);
        acc = fmaf(w, w, acc);
    }

    __shared__ float red[256];
    red[t] = acc;
    __syncthreads();
    for (int s = 128; s > 0; s >>= 1) {
        if (t < s) red[t] += red[t + s];
        __syncthreads();
    }
    if (t == 0) mns[o] = mag[o] / sqrtf(red[0]);
}

// ---------------------------------------------------------------------------
// Kernel 4: segment-dependent epilogue on out.
// Grid: (OUT_DIM/256, M_TOK), one thread per (m, o).
// ---------------------------------------------------------------------------
__global__ __launch_bounds__(256)
void adapter_epilogue(float* __restrict__ out,
                      const float* __restrict__ T,
                      const float* __restrict__ l1b, const float* __restrict__ l2b,
                      const float* __restrict__ vb,  const float* __restrict__ dbm,
                      const float* __restrict__ b_vec,
                      const float* __restrict__ mns)
{
    const int m = blockIdx.y;
    const int o = blockIdx.x * 256 + threadIdx.x;
    const int seg = m >> 11;

    __shared__ float st[RANK];
    if (threadIdx.x < RANK) st[threadIdx.x] = T[(size_t)m * RANK + threadIdx.x];
    __syncthreads();

    const float* B = (seg == 0) ? l1b : (seg == 1) ? l2b : (seg == 2) ? vb : dbm;

    float dot = 0.0f;
#pragma unroll
    for (int q = 0; q < 4; ++q) {
        floatx4 bv = *(const floatx4*)(B + (size_t)o * RANK + q * 4);
#pragma unroll
        for (int e = 0; e < 4; ++e)
            dot = fmaf(st[q * 4 + e], bv[e], dot);
    }

    const size_t idx = (size_t)m * OUT_DIM + o;
    if (seg < 2) {
        out[idx] += 2.0f * dot;                       // LoRA scaling
    } else if (seg == 2) {
        out[idx] += dot * b_vec[o];                   // VeRA (d folded upstream)
    } else {
        float mv = mns[o];
        float r0 = out[idx];
        out[idx] = r0 * (mv - 1.0f) + mv * 2.0f * dot; // DoRA index_copy form
    }
}

// ---------------------------------------------------------------------------
// Host-side launcher
// ---------------------------------------------------------------------------
extern "C" void kernel_launch(void* const* d_in, const int* in_sizes, int n_in,
                              void* d_out, int out_size, void* d_ws, size_t ws_size,
                              hipStream_t stream)
{
    const float* data   = (const float*)d_in[0];
    const float* W      = (const float*)d_in[1];
    const float* l1a    = (const float*)d_in[2];
    const float* l1b    = (const float*)d_in[3];
    const float* l2a    = (const float*)d_in[4];
    const float* l2b    = (const float*)d_in[5];
    const float* va     = (const float*)d_in[6];
    const float* vb     = (const float*)d_in[7];
    const float* d_vec  = (const float*)d_in[8];
    const float* b_vec  = (const float*)d_in[9];
    const float* da     = (const float*)d_in[10];
    const float* db     = (const float*)d_in[11];
    const float* mag    = (const float*)d_in[12];
    float* out = (float*)d_out;

    // workspace: Whi (32MB) | Wlo (32MB) | T (512KB) | mns (16KB)
    char* ws = (char*)d_ws;
    __bf16* Whi = (__bf16*)ws;
    __bf16* Wlo = Whi + (size_t)OUT_DIM * IN_DIM;
    float*  T   = (float*)(Wlo + (size_t)OUT_DIM * IN_DIM);
    float*  mns = T + (size_t)M_TOK * RANK;

    // 0) one-time W split (W is reused 64x by the GEMM sweeps)
    split_w<<<(OUT_DIM * IN_DIM / 4) / 256, 256, 0, stream>>>(W, Whi, Wlo);

    // 1) main GEMM (ping-pong LDS, async B staging)
    dim3 gGemm(OUT_DIM / BN, M_TOK / BM);      // (32, 64)
    gemm_bf16x3<<<gGemm, 256, SMEM_BYTES, stream>>>(data, Whi, Wlo, out);

    // 2) adapter projections + DoRA norms
    adapter_proj<<<M_TOK, 256, 0, stream>>>(data, l1a, l2a, va, da, d_vec, T);
    dora_norm<<<OUT_DIM, 256, 0, stream>>>(W, da, db, mag, mns);

    // 3) segment epilogue
    dim3 gEpi(OUT_DIM / 256, M_TOK);           // (16, 8192)
    adapter_epilogue<<<gEpi, 256, 0, stream>>>(out, T, l1b, l2b, vb, db, b_vec, mns);
}
